// CausalSelfAttention_8796093022589
// MI455X (gfx1250) — matrix operand order repaired and checked by reading, not compiled, not run on hardware
//
#include <hip/hip_runtime.h>

#define Bn 4
#define Tn 2048
#define Dn 1024
#define Hn 16
#define HDn 64

typedef __attribute__((ext_vector_type(16))) _Float16 v16h;
typedef __attribute__((ext_vector_type(8)))  _Float16 v8h;
typedef __attribute__((ext_vector_type(8)))  float    v8f;
typedef int v4i __attribute__((__vector_size__(16)));

union V16 { v16h v; v8h h[2]; };

__device__ __forceinline__ v8f wmma16(v16h a, v16h b, v8f c) {
    // D = A(16x32 f16) * B(32x16 f16) + C(16x16 f32)
    return __builtin_amdgcn_wmma_f32_16x16x32_f16(
        /*neg_a=*/false, a, /*neg_b=*/false, b,
        /*c_mod=*/(short)0, c, /*reuse_a=*/false, /*reuse_b=*/false);
}

// ---- CDNA5 async global->LDS (ASYNCcnt-tracked), 16B per lane ----
__device__ __forceinline__ void async_copy_b128(void* lds_dst, const void* gsrc) {
#if __has_builtin(__builtin_amdgcn_global_load_async_to_lds_b128)
    __builtin_amdgcn_global_load_async_to_lds_b128(
        (__attribute__((address_space(1))) v4i*)gsrc,
        (__attribute__((address_space(3))) v4i*)lds_dst,
        /*offset=*/0, /*cpol=*/0);
#else
    unsigned int loff = (unsigned int)(unsigned long long)lds_dst;
    asm volatile("global_load_async_to_lds_b128 %0, %1, off"
                 :: "v"(loff), "v"(gsrc) : "memory");
#endif
}

__device__ __forceinline__ void wait_async0() {
#if __has_builtin(__builtin_amdgcn_s_wait_asynccnt)
    __builtin_amdgcn_s_wait_asynccnt(0);
#else
    asm volatile("s_wait_asynccnt 0" ::: "memory");
#endif
}

// ---------------- conversion kernels ----------------

__global__ void cvt_f32_to_f16(const float* __restrict__ in,
                               _Float16* __restrict__ out, int n) {
    int i = blockIdx.x * blockDim.x + threadIdx.x;
    if (i < n) out[i] = (_Float16)in[i];
}

// in: [K][N] fp32 row-major  ->  out: [N][K] f16 row-major
__global__ void tcvt_f32_to_f16(const float* __restrict__ in,
                                _Float16* __restrict__ out, int K, int N) {
    int i = blockIdx.x * blockDim.x + threadIdx.x;
    if (i < K * N) {
        int nn = i / K, k = i % K;
        out[i] = (_Float16)in[(size_t)k * N + nn];
    }
}

// ---------------- QKV projection GEMM (register-blocked 32x64/wave) ----------------
// C[8192,3072] = X[8192,1024] * WqkvT[3072,1024]^T + bias
// scatter epilogue into Q[b,h,t,hd], K[b,h,t,hd], Vt[b,h,hd,t]  (f16)

__global__ __launch_bounds__(256) void qkv_gemm(
    const _Float16* __restrict__ X, const _Float16* __restrict__ Wt,
    const float* __restrict__ bias, _Float16* __restrict__ Q,
    _Float16* __restrict__ Kd, _Float16* __restrict__ Vt) {
    const int lane = threadIdx.x & 31, wid = threadIdx.x >> 5;
    const int half = lane >> 4, loc = lane & 15;
    const int tile = blockIdx.x * 8 + wid;
    const int NT = (3 * Dn) / 64;            // 48 col blocks
    const int tn = tile % NT, tm = tile / NT;
    const int row0 = tm * 32, col0 = tn * 64;

    const _Float16* arow0 = X + (size_t)(row0 + loc) * Dn;
    const _Float16* arow1 = X + (size_t)(row0 + 16 + loc) * Dn;
    const _Float16* brow0 = Wt + (size_t)(col0 + loc) * Dn;
    const _Float16* brow1 = Wt + (size_t)(col0 + 16 + loc) * Dn;
    const _Float16* brow2 = Wt + (size_t)(col0 + 32 + loc) * Dn;
    const _Float16* brow3 = Wt + (size_t)(col0 + 48 + loc) * Dn;

    v8f acc[2][4];
#pragma unroll
    for (int mi = 0; mi < 2; ++mi)
#pragma unroll
        for (int ni = 0; ni < 4; ++ni)
            acc[mi][ni] = (v8f){0,0,0,0,0,0,0,0};

    for (int kc = 0; kc < Dn; kc += 32) {
        V16 a0, a1;
        a0.h[0] = *(const v8h*)(arow0 + kc + half * 16);
        a0.h[1] = *(const v8h*)(arow0 + kc + 8 + half * 16);
        a1.h[0] = *(const v8h*)(arow1 + kc + half * 16);
        a1.h[1] = *(const v8h*)(arow1 + kc + 8 + half * 16);
        v16h b0 = *(const v16h*)(brow0 + kc + half * 16);
        v16h b1 = *(const v16h*)(brow1 + kc + half * 16);
        v16h b2 = *(const v16h*)(brow2 + kc + half * 16);
        v16h b3 = *(const v16h*)(brow3 + kc + half * 16);
        acc[0][0] = wmma16(a0.v, b0, acc[0][0]);
        acc[0][1] = wmma16(a0.v, b1, acc[0][1]);
        acc[0][2] = wmma16(a0.v, b2, acc[0][2]);
        acc[0][3] = wmma16(a0.v, b3, acc[0][3]);
        acc[1][0] = wmma16(a1.v, b0, acc[1][0]);
        acc[1][1] = wmma16(a1.v, b1, acc[1][1]);
        acc[1][2] = wmma16(a1.v, b2, acc[1][2]);
        acc[1][3] = wmma16(a1.v, b3, acc[1][3]);
    }

#pragma unroll
    for (int ni = 0; ni < 4; ++ni) {
        const int n = col0 + ni * 16 + loc;   // global column [0,3072)
        const float bn = bias[n];
        const int which = n / Dn;             // 0=q, 1=k, 2=v (uniform per 64-col block)
        const int dd = n % Dn;
        const int h = dd / HDn, hd = dd % HDn;
#pragma unroll
        for (int mi = 0; mi < 2; ++mi) {
#pragma unroll
            for (int r = 0; r < 8; ++r) {
                const int m = row0 + mi * 16 + half * 8 + r;
                const int b_ = m / Tn, t = m % Tn;
                const _Float16 hv = (_Float16)(acc[mi][ni][r] + bn);
                const size_t bh = (size_t)(b_ * Hn + h);
                if (which == 0)      Q [(bh * Tn + t) * HDn + hd] = hv;
                else if (which == 1) Kd[(bh * Tn + t) * HDn + hd] = hv;
                else                 Vt[(bh * HDn + hd) * Tn + t] = hv;
            }
        }
    }
}

// ---------------- flash attention (block-cooperative K/V staging) ----------------
// block = 8 waves = 128 q rows of one (b,h); K/V chunk staged once per block via
// async global->LDS; each wave owns a 16-row q tile with online softmax.

__global__ __launch_bounds__(256) void attn_kernel(
    const _Float16* __restrict__ Q, const _Float16* __restrict__ Kd,
    const _Float16* __restrict__ Vt, _Float16* __restrict__ AO) {
    __shared__ alignas(128) _Float16 Kt[32][64];    // [key][hd]   4KB
    __shared__ alignas(128) _Float16 Vl[64][32];    // [hd][key]   4KB
    __shared__ alignas(64)  _Float16 Pl[8][16][32]; // per-wave P transpose 8KB

    const int tid = threadIdx.x;
    const int lane = tid & 31, wid = tid >> 5;
    const int half = lane >> 4, loc = lane & 15;

    const int nblk = Tn / 128;                       // 16 q-blocks per (b,h)
    const int bh = blockIdx.x / nblk;
    const int q0b = (blockIdx.x % nblk) * 128;       // block's first q row
    const int q0 = q0b + wid * 16;                   // this wave's q tile

    const _Float16* Qb = Q  + (size_t)bh * Tn * HDn;
    const _Float16* Kb = Kd + (size_t)bh * Tn * HDn;
    const _Float16* Vb = Vt + (size_t)bh * HDn * Tn;

    // staging addresses for this thread (16B per async op, 2 ops/thread/chunk)
    const int kkey = tid >> 3, kseg = tid & 7;       // K: 32 rows x 8 segs
    const int vd = tid >> 2, vseg = tid & 3;         // V: 64 rows x 4 segs
    _Float16* ldsK = &Kt[kkey][kseg * 8];
    _Float16* ldsV = &Vl[vd][vseg * 8];

    // Q A-fragments for hd chunks [0,32) and [32,64)
    const _Float16* qrow = Qb + (size_t)(q0 + loc) * HDn;
    V16 aq0, aq1;
    aq0.h[0] = *(const v8h*)(qrow + half * 16);
    aq0.h[1] = *(const v8h*)(qrow + 8 + half * 16);
    aq1.h[0] = *(const v8h*)(qrow + 32 + half * 16);
    aq1.h[1] = *(const v8h*)(qrow + 40 + half * 16);

    float mrow[8], lrow[8];
    v8f o0 = {0,0,0,0,0,0,0,0}, o1 = {0,0,0,0,0,0,0,0};
    v8f o2 = {0,0,0,0,0,0,0,0}, o3 = {0,0,0,0,0,0,0,0};
#pragma unroll
    for (int r = 0; r < 8; ++r) { mrow[r] = -1e30f; lrow[r] = 0.f; }

    const int kcmax_blk = (q0b + 127) / 32;          // block-uniform bound
    const int kcmax_me  = (q0 + 15) / 32;            // wave-uniform bound

    for (int kc = 0; kc <= kcmax_blk; ++kc) {
        const int n0 = kc * 32;
        // ---- stage K[32][64] and V^T[64][32] chunk into LDS (async) ----
        async_copy_b128(ldsK, Kb + (size_t)(n0 + kkey) * HDn + kseg * 8);
        async_copy_b128(ldsV, Vb + (size_t)vd * Tn + n0 + vseg * 8);
        wait_async0();
        __syncthreads();

        if (kc <= kcmax_me) {                        // wave-uniform: EXEC stays full
            // ---- S = Q K^T from LDS ----
            v16h b00 = *(const v16h*)(&Kt[loc][half * 16]);
            v16h b01 = *(const v16h*)(&Kt[loc][32 + half * 16]);
            v16h b10 = *(const v16h*)(&Kt[16 + loc][half * 16]);
            v16h b11 = *(const v16h*)(&Kt[16 + loc][32 + half * 16]);
            v8f s0 = {0,0,0,0,0,0,0,0}, s1 = {0,0,0,0,0,0,0,0};
            s0 = wmma16(aq0.v, b00, s0); s0 = wmma16(aq1.v, b01, s0);
            s1 = wmma16(aq0.v, b10, s1); s1 = wmma16(aq1.v, b11, s1);

            // ---- online softmax (row = half*8+r, cols loc / 16+loc) ----
#pragma unroll
            for (int r = 0; r < 8; ++r) {
                const int rowq = q0 + half * 8 + r;
                const int key0 = n0 + loc, key1 = n0 + 16 + loc;
                float f0 = (key0 <= rowq) ? s0[r] * 0.125f : -1e30f;
                float f1 = (key1 <= rowq) ? s1[r] * 0.125f : -1e30f;
                float mx = fmaxf(f0, f1);
                mx = fmaxf(mx, __shfl_xor(mx, 1, 32));
                mx = fmaxf(mx, __shfl_xor(mx, 2, 32));
                mx = fmaxf(mx, __shfl_xor(mx, 4, 32));
                mx = fmaxf(mx, __shfl_xor(mx, 8, 32));
                const float mnew = fmaxf(mrow[r], mx);
                const float p0 = __expf(f0 - mnew);
                const float p1 = __expf(f1 - mnew);
                float ps = p0 + p1;
                ps += __shfl_xor(ps, 1, 32);
                ps += __shfl_xor(ps, 2, 32);
                ps += __shfl_xor(ps, 4, 32);
                ps += __shfl_xor(ps, 8, 32);
                const float alpha = __expf(mrow[r] - mnew);
                lrow[r] = lrow[r] * alpha + ps;
                mrow[r] = mnew;
                o0[r] *= alpha; o1[r] *= alpha; o2[r] *= alpha; o3[r] *= alpha;
                Pl[wid][half * 8 + r][loc]      = (_Float16)p0;
                Pl[wid][half * 8 + r][16 + loc] = (_Float16)p1;
            }

            // ---- P C-layout -> A-layout via LDS (wave-private, DS in-order) ----
            V16 ap;
            ap.h[0] = *(const v8h*)(&Pl[wid][loc][half * 16]);
            ap.h[1] = *(const v8h*)(&Pl[wid][loc][8 + half * 16]);

            // ---- O += P V from LDS (V^T rows contiguous over keys) ----
            v16h bv0 = *(const v16h*)(&Vl[0  + loc][half * 16]);
            v16h bv1 = *(const v16h*)(&Vl[16 + loc][half * 16]);
            v16h bv2 = *(const v16h*)(&Vl[32 + loc][half * 16]);
            v16h bv3 = *(const v16h*)(&Vl[48 + loc][half * 16]);
            o0 = wmma16(ap.v, bv0, o0);
            o1 = wmma16(ap.v, bv1, o1);
            o2 = wmma16(ap.v, bv2, o2);
            o3 = wmma16(ap.v, bv3, o3);
        }
        __syncthreads();                             // protect Kt/Vl before next stage
    }

    // ---- epilogue: O / l, write [B,T,D] f16 ----
    const int b_ = bh / Hn, h = bh % Hn;
#pragma unroll
    for (int r = 0; r < 8; ++r) {
        const int t = q0 + half * 8 + r;
        const float inv = 1.f / lrow[r];
        const size_t base = ((size_t)(b_ * Tn + t)) * Dn + h * HDn;
        AO[base + 0  + loc] = (_Float16)(o0[r] * inv);
        AO[base + 16 + loc] = (_Float16)(o1[r] * inv);
        AO[base + 32 + loc] = (_Float16)(o2[r] * inv);
        AO[base + 48 + loc] = (_Float16)(o3[r] * inv);
    }
}

// ---------------- output projection GEMM (register-blocked 32x64/wave) ----------------
// out[8192,1024] fp32 = AO[8192,1024] * WprojT[1024,1024]^T + bias

__global__ __launch_bounds__(256) void proj_gemm(
    const _Float16* __restrict__ A, const _Float16* __restrict__ Wt,
    const float* __restrict__ bias, float* __restrict__ out) {
    const int lane = threadIdx.x & 31, wid = threadIdx.x >> 5;
    const int half = lane >> 4, loc = lane & 15;
    const int tile = blockIdx.x * 8 + wid;
    const int NT = Dn / 64;                  // 16 col blocks
    const int tn = tile % NT, tm = tile / NT;
    const int row0 = tm * 32, col0 = tn * 64;

    const _Float16* arow0 = A + (size_t)(row0 + loc) * Dn;
    const _Float16* arow1 = A + (size_t)(row0 + 16 + loc) * Dn;
    const _Float16* brow0 = Wt + (size_t)(col0 + loc) * Dn;
    const _Float16* brow1 = Wt + (size_t)(col0 + 16 + loc) * Dn;
    const _Float16* brow2 = Wt + (size_t)(col0 + 32 + loc) * Dn;
    const _Float16* brow3 = Wt + (size_t)(col0 + 48 + loc) * Dn;

    v8f acc[2][4];
#pragma unroll
    for (int mi = 0; mi < 2; ++mi)
#pragma unroll
        for (int ni = 0; ni < 4; ++ni)
            acc[mi][ni] = (v8f){0,0,0,0,0,0,0,0};

    for (int kc = 0; kc < Dn; kc += 32) {
        V16 a0, a1;
        a0.h[0] = *(const v8h*)(arow0 + kc + half * 16);
        a0.h[1] = *(const v8h*)(arow0 + kc + 8 + half * 16);
        a1.h[0] = *(const v8h*)(arow1 + kc + half * 16);
        a1.h[1] = *(const v8h*)(arow1 + kc + 8 + half * 16);
        v16h b0 = *(const v16h*)(brow0 + kc + half * 16);
        v16h b1 = *(const v16h*)(brow1 + kc + half * 16);
        v16h b2 = *(const v16h*)(brow2 + kc + half * 16);
        v16h b3 = *(const v16h*)(brow3 + kc + half * 16);
        acc[0][0] = wmma16(a0.v, b0, acc[0][0]);
        acc[0][1] = wmma16(a0.v, b1, acc[0][1]);
        acc[0][2] = wmma16(a0.v, b2, acc[0][2]);
        acc[0][3] = wmma16(a0.v, b3, acc[0][3]);
        acc[1][0] = wmma16(a1.v, b0, acc[1][0]);
        acc[1][1] = wmma16(a1.v, b1, acc[1][1]);
        acc[1][2] = wmma16(a1.v, b2, acc[1][2]);
        acc[1][3] = wmma16(a1.v, b3, acc[1][3]);
    }

#pragma unroll
    for (int ni = 0; ni < 4; ++ni) {
        const int n = col0 + ni * 16 + loc;
        const float bn = bias[n];
#pragma unroll
        for (int mi = 0; mi < 2; ++mi) {
#pragma unroll
            for (int r = 0; r < 8; ++r) {
                const int m = row0 + mi * 16 + half * 8 + r;
                out[(size_t)m * Dn + n] = acc[mi][ni][r] + bn;
            }
        }
    }
}

// ---------------- host-side launch ----------------

extern "C" void kernel_launch(void* const* d_in, const int* in_sizes, int n_in,
                              void* d_out, int out_size, void* d_ws, size_t ws_size,
                              hipStream_t stream) {
    const float* x      = (const float*)d_in[0];
    const float* w_qkv  = (const float*)d_in[1];
    const float* b_qkv  = (const float*)d_in[2];
    const float* w_proj = (const float*)d_in[3];
    const float* b_proj = (const float*)d_in[4];
    float* out = (float*)d_out;

    _Float16* ws = (_Float16*)d_ws;
    size_t off = 0;
    const size_t NX   = (size_t)Bn * Tn * Dn;       // 8,388,608
    const size_t NWQ  = (size_t)3 * Dn * Dn;        // 3,145,728
    const size_t NWP  = (size_t)Dn * Dn;            // 1,048,576
    const size_t NQKV = (size_t)Bn * Hn * Tn * HDn; // 8,388,608

    _Float16* xh     = ws + off; off += NX;
    _Float16* wqkvT  = ws + off; off += NWQ;
    _Float16* wprojT = ws + off; off += NWP;
    _Float16* Qh     = ws + off; off += NQKV;
    _Float16* Kh     = ws + off; off += NQKV;
    _Float16* Vth    = ws + off; off += NQKV;
    _Float16* aoh    = ws + off; off += NX;

    cvt_f32_to_f16<<<(int)((NX + 255) / 256), 256, 0, stream>>>(x, xh, (int)NX);
    tcvt_f32_to_f16<<<(int)((NWQ + 255) / 256), 256, 0, stream>>>(w_qkv, wqkvT, Dn, 3 * Dn);
    tcvt_f32_to_f16<<<(int)((NWP + 255) / 256), 256, 0, stream>>>(w_proj, wprojT, Dn, Dn);

    // (8192/32)*(3072/64) = 12288 wave-blocks, 8 waves/block
    qkv_gemm<<<1536, 256, 0, stream>>>(xh, wqkvT, b_qkv, Qh, Kh, Vth);

    // B*H*(T/128) = 1024 blocks, 8 waves each
    attn_kernel<<<1024, 256, 0, stream>>>(Qh, Kh, Vth, aoh);

    // (8192/32)*(1024/64) = 4096 wave-blocks, 8 waves/block
    proj_gemm<<<512, 256, 0, stream>>>(aoh, wprojT, b_proj, out);
}
